// trans_encoder_38835094290463
// MI455X (gfx1250) — matrix operand order repaired and checked by reading, not compiled, not run on hardware
//
#include <hip/hip_runtime.h>
#include <hip/hip_bf16.h>

// Point Cloud Transformer encoder for MI455X (gfx1250), bf16 WMMA + fp32 accum.
// P = B*N = 32768 points. All feature maps stored channel-major [C][P] fp32.

#define PTS  32768
#define NPTS 2048
#define NBAT 16
#define LDS_STRIDE 48  // halfs per staged column (32 data + 16 pad; keeps 32B align)

typedef __attribute__((ext_vector_type(16))) __bf16 v16bf;
typedef __attribute__((ext_vector_type(8)))  __bf16 v8bf;
typedef __attribute__((ext_vector_type(8)))  float  v8f;
typedef __attribute__((ext_vector_type(4)))  float  v4f;

__device__ inline v8f wmma_bf16(v16bf a, v16bf b, v8f c) {
  return __builtin_amdgcn_wmma_f32_16x16x32_bf16(false, a, false, b, (short)0, c,
                                                 false, false);
}

__device__ inline void atomicMaxF(float* addr, float val) {
  int* ia = reinterpret_cast<int*>(addr);
  int cur = *ia;
  while (__int_as_float(cur) < val) {
    int old = atomicCAS(ia, cur, __float_as_int(val));
    if (old == cur) break;
    cur = old;
  }
}

#define ACCJ(a0, a1, a2, a3, j, r) \
  ((j) == 0 ? a0[r] : (j) == 1 ? a1[r] : (j) == 2 ? a2[r] : a3[r])
#define ACC_AT(mt, j, r) \
  ((mt) == 0 ? ACCJ(acc00, acc01, acc02, acc03, j, r) \
             : ACCJ(acc10, acc11, acc12, acc13, j, r))

// Generic WMMA GEMM: Y[m][col] = epilogue( sum_k A[m][k] * (B[k][col] - Bsub?) ).
// Block: 256 threads = 8 waves; block tile = 32(M) x 512(cols); wave owns 32x64
// via 8 WMMAs per K-step (2 A fragments x 4 B fragments -> LDS traffic amortized
// over twice the matrix ops). B panel (32k x 512cols) staged to LDS as bf16 in
// [col][k] order: each lane's B fragment is two 16B runs (2x ds_load_b128).
// Double-buffered -> one barrier per K-step. grid = (Ncols/512, M/32).
// MODE 0: +bias                      MODE 1: relu(+bias)
// MODE 2: resid + relu(g*(+bias)*rsqrt(1+eps) + be)   (SA layer output)
// MODE 3: acc / (1e-9 + colsum[col])                  (offset-attn renorm)
// MODE 4: per-batch max over cols -> atomicMax(outmax[batch*M+m]) (fused final max)
template <int MODE, bool TRANSA, bool HASSUB, bool MASKK>
__global__ __launch_bounds__(256) void gemm_wmma_k(
    const float* __restrict__ A, int lda,
    const float* __restrict__ Bm, int ldb, const float* __restrict__ Bsub,
    float* __restrict__ Y, int ldy, int M, int K, int Ncols,
    const float* __restrict__ bias,
    const float* __restrict__ g, const float* __restrict__ be,
    const float* __restrict__ resid, const float* __restrict__ colsum,
    float* __restrict__ outmax, int colsPerBatch) {
  __shared__ __bf16 lbs[2][512 * LDS_STRIDE];  // 2 x 48 KB

  const int t     = threadIdx.x;
  const int lane  = t & 31;
  const int wave  = t >> 5;
  const int hi    = lane >> 4;
  const int lo    = lane & 15;
  const int otile = blockIdx.y << 5;  // 32 output rows per block
  const int cbase = blockIdx.x * 512;

  const v8f vz = {0.f, 0.f, 0.f, 0.f, 0.f, 0.f, 0.f, 0.f};
  v8f acc00 = vz, acc01 = vz, acc02 = vz, acc03 = vz;
  v8f acc10 = vz, acc11 = vz, acc12 = vz, acc13 = vz;

  int bufi = 0;
  for (int kb = 0; kb < K; kb += 32) {
    if (kb + 32 < K)  // pull next panel toward L2 (global_prefetch_b8)
      __builtin_prefetch(&Bm[(kb + 32) * ldb + cbase + t * 2], 0, 1);

    // ---- cooperative stage: B panel [32 x 512] fp32 -> LDS [col][k] bf16 ----
    __bf16* dst = lbs[bufi];
    for (int half = 0; half < 2; ++half) {
      const int col  = half * 256 + t;
      const int gcol = cbase + col;
      for (int kk8 = 0; kk8 < 32; kk8 += 8) {
        v8bf pk;
#pragma unroll
        for (int u = 0; u < 8; ++u) {
          const int k = kb + kk8 + u;
          float v;
          if (MASKK) {
            const int kc = k < K ? k : K - 1;  // clamped in-bounds load + mask
            v = Bm[kc * ldb + gcol];
            if (k >= K) v = 0.f;
          } else {
            v = Bm[k * ldb + gcol];
          }
          if (HASSUB) v -= Bsub[k * ldb + gcol];  // (HASSUB never paired w/ MASKK)
          pk[u] = (__bf16)v;
        }
        *reinterpret_cast<v8bf*>(&dst[col * LDS_STRIDE + kk8]) = pk;  // ds_store_b128
      }
    }

    // ---- two A fragments: 16x32 bf16 each. ISA layout: lane half hi,
    //      element e -> k = (e>>3)*16 + hi*8 + (e&7) (two contiguous 8-runs) ----
    v16bf a0, a1;
    if (TRANSA) {  // A[k][m] column-major (energy: A = q^T); coalesced across lo
#pragma unroll
      for (int e = 0; e < 16; ++e) {
        const int k = kb + ((e >> 3) << 4) + (hi << 3) + (e & 7);
        a0[e] = (__bf16)A[k * lda + otile + lo];
        a1[e] = (__bf16)A[k * lda + otile + 16 + lo];
      }
    } else if (MASKK) {
#pragma unroll
      for (int e = 0; e < 16; ++e) {
        const int k  = kb + ((e >> 3) << 4) + (hi << 3) + (e & 7);
        const int kc = k < K ? k : K - 1;
        const float u0 = A[(otile + lo) * lda + kc];
        const float u1 = A[(otile + 16 + lo) * lda + kc];
        a0[e] = (__bf16)(k < K ? u0 : 0.f);
        a1[e] = (__bf16)(k < K ? u1 : 0.f);
      }
    } else {  // vectorized: 8x global_load_b128
      const float* ap0 = A + (otile + lo) * lda + kb + hi * 8;
      const float* ap1 = ap0 + 16 * lda;
      const v4f f0 = *reinterpret_cast<const v4f*>(ap0 + 0);
      const v4f f1 = *reinterpret_cast<const v4f*>(ap0 + 4);
      const v4f f2 = *reinterpret_cast<const v4f*>(ap0 + 16);
      const v4f f3 = *reinterpret_cast<const v4f*>(ap0 + 20);
      const v4f h0 = *reinterpret_cast<const v4f*>(ap1 + 0);
      const v4f h1 = *reinterpret_cast<const v4f*>(ap1 + 4);
      const v4f h2 = *reinterpret_cast<const v4f*>(ap1 + 16);
      const v4f h3 = *reinterpret_cast<const v4f*>(ap1 + 20);
#pragma unroll
      for (int i = 0; i < 4; ++i) {
        a0[i]      = (__bf16)f0[i];
        a0[4 + i]  = (__bf16)f1[i];
        a0[8 + i]  = (__bf16)f2[i];
        a0[12 + i] = (__bf16)f3[i];
        a1[i]      = (__bf16)h0[i];
        a1[4 + i]  = (__bf16)h1[i];
        a1[8 + i]  = (__bf16)h2[i];
        a1[12 + i] = (__bf16)h3[i];
      }
    }

    __syncthreads();  // staging visible to whole block; also guards buffer reuse

    // ---- B fragments: two 16B runs per lane (k = hi*8.., 16+hi*8..) -> 2x ds_load_b128 each ----
    const __bf16* src = lbs[bufi] + (wave * 64 + lo) * LDS_STRIDE + hi * 8;
#define LDB16(p) __builtin_shufflevector(*reinterpret_cast<const v8bf*>(p), *reinterpret_cast<const v8bf*>((p) + 16), 0, 1, 2, 3, 4, 5, 6, 7, 8, 9, 10, 11, 12, 13, 14, 15)
    const v16bf b0 = LDB16(src + 0 * 16 * LDS_STRIDE);
    const v16bf b1 = LDB16(src + 1 * 16 * LDS_STRIDE);
    const v16bf b2 = LDB16(src + 2 * 16 * LDS_STRIDE);
    const v16bf b3 = LDB16(src + 3 * 16 * LDS_STRIDE);

    acc00 = wmma_bf16(a0, b0, acc00);
    acc10 = wmma_bf16(a1, b0, acc10);
    acc01 = wmma_bf16(a0, b1, acc01);
    acc11 = wmma_bf16(a1, b1, acc11);
    acc02 = wmma_bf16(a0, b2, acc02);
    acc12 = wmma_bf16(a1, b2, acc12);
    acc03 = wmma_bf16(a0, b3, acc03);
    acc13 = wmma_bf16(a1, b3, acc13);

    bufi ^= 1;
  }

  // ---- epilogue (C/D layout: n = lane&15, m = r + 8*hi) ----
  const float bninv = rsqrtf(1.0f + 1e-5f);
  const int pwave = cbase + wave * 64;
#pragma unroll
  for (int mt = 0; mt < 2; ++mt) {
#pragma unroll
    for (int r = 0; r < 8; ++r) {
      const int m = otile + mt * 16 + r + (hi << 3);
      const float bi = bias ? bias[m] : 0.f;
      if (MODE == 4) {
        float mx = -1e30f;
#pragma unroll
        for (int j = 0; j < 4; ++j) mx = fmaxf(mx, ACC_AT(mt, j, r) + bi);
#pragma unroll
        for (int s = 1; s < 16; s <<= 1) mx = fmaxf(mx, __shfl_xor(mx, s, 32));
        if (lo == 0)  // wave's 64 cols sit inside one batch (64 | 2048)
          atomicMaxF(&outmax[(pwave / colsPerBatch) * M + m], mx);
      } else {
#pragma unroll
        for (int j = 0; j < 4; ++j) {
          const int pcol = pwave + j * 16 + lo;
          float val = ACC_AT(mt, j, r) + bi;
          if (MODE == 1) val = fmaxf(val, 0.f);
          if (MODE == 2)
            val = resid[m * ldy + pcol] + fmaxf(g[m] * val * bninv + be[m], 0.f);
          if (MODE == 3) val = ACC_AT(mt, j, r) / (1e-9f + colsum[pcol]);
          Y[m * ldy + pcol] = val;
        }
      }
    }
  }
}

// --- elementwise / reduction helpers -------------------------------------

__global__ void transpose_in_k(const float* __restrict__ in, float* __restrict__ x0) {
  int p = blockIdx.x * 256 + threadIdx.x;  // [0, PTS)
  float a = in[p * 3 + 0], b = in[p * 3 + 1], c = in[p * 3 + 2];
  x0[0 * PTS + p] = a; x0[1 * PTS + p] = b; x0[2 * PTS + p] = c;
}

__global__ __launch_bounds__(256) void rowmax_k(const float* __restrict__ lf,
                                                float* __restrict__ gmax) {
  int wave = threadIdx.x >> 5, lane = threadIdx.x & 31;
  int row = blockIdx.x * 8 + wave;  // b*128 + c, row < 2048
  int b = row >> 7, c = row & 127;
  const float* p = lf + c * PTS + b * NPTS;
  float m = -1e30f;
  for (int n = lane; n < NPTS; n += 32) m = fmaxf(m, p[n]);
#pragma unroll
  for (int s = 16; s > 0; s >>= 1) m = fmaxf(m, __shfl_xor(m, s, 32));
  if (lane == 0) gmax[row] = m;
}

__global__ void build_totalf_k(const float* __restrict__ lf,
                               const float* __restrict__ gmax,
                               float* __restrict__ tf) {
  int p = blockIdx.x * 256 + threadIdx.x;
  int c = blockIdx.y;
  int b = p >> 11;
  tf[c * PTS + p]         = lf[c * PTS + p];
  tf[(128 + c) * PTS + p] = gmax[b * 128 + c];
}

__global__ void add_k(const float* __restrict__ a, const float* __restrict__ b,
                      float* __restrict__ y) {
  int p = blockIdx.x * 256 + threadIdx.x;
  int i = blockIdx.y * PTS + p;
  y[i] = a[i] + b[i];
}

// One block per attention row; softmax over 2048 columns in-place.
__global__ __launch_bounds__(256) void softmax_rows_k(float* __restrict__ attn) {
  float* rp = attn + blockIdx.x * NPTS;
  const int t = threadIdx.x;
  float vals[8];
  float mx = -1e30f;
#pragma unroll
  for (int i = 0; i < 8; i++) { vals[i] = rp[t + i * 256]; mx = fmaxf(mx, vals[i]); }
  __shared__ float sh[8];
#pragma unroll
  for (int s = 16; s > 0; s >>= 1) mx = fmaxf(mx, __shfl_xor(mx, s, 32));
  if ((t & 31) == 0) sh[t >> 5] = mx;
  __syncthreads();
  mx = sh[0];
#pragma unroll
  for (int w = 1; w < 8; w++) mx = fmaxf(mx, sh[w]);
  float sum = 0.f;
#pragma unroll
  for (int i = 0; i < 8; i++) { vals[i] = __expf(vals[i] - mx); sum += vals[i]; }
#pragma unroll
  for (int s = 16; s > 0; s >>= 1) sum += __shfl_xor(sum, s, 32);
  __syncthreads();
  if ((t & 31) == 0) sh[t >> 5] = sum;
  __syncthreads();
  sum = 0.f;
#pragma unroll
  for (int w = 0; w < 8; w++) sum += sh[w];
  const float inv = 1.0f / sum;
#pragma unroll
  for (int i = 0; i < 8; i++) rp[t + i * 256] = vals[i] * inv;
}

__global__ void colsum_k(const float* __restrict__ attn, float* __restrict__ cs) {
  int m = blockIdx.x * 256 + threadIdx.x;  // coalesced across threads
  float s = 0.f;
  for (int n = 0; n < NPTS; n++) s += attn[n * NPTS + m];
  cs[m] = s;
}

__global__ void init_out_k(float* out) { out[blockIdx.x * 256 + threadIdx.x] = -1e30f; }

// --- host-side driver ----------------------------------------------------

template <int MODE, bool TRANSA = false, bool HASSUB = false, bool MASKK = false>
static inline void gemm(hipStream_t s, const float* A, int lda,
                        const float* Bm, int ldb, const float* Bsub,
                        float* Y, int ldy, int M, int K, int Ncols,
                        const float* bias,
                        const float* g = nullptr, const float* be = nullptr,
                        const float* resid = nullptr, const float* colsum = nullptr,
                        float* outmax = nullptr, int colsPerBatch = 1) {
  dim3 grid(Ncols / 512, M / 32);
  gemm_wmma_k<MODE, TRANSA, HASSUB, MASKK><<<grid, 256, 0, s>>>(
      A, lda, Bm, ldb, Bsub, Y, ldy, M, K, Ncols, bias, g, be, resid, colsum,
      outmax, colsPerBatch);
}

extern "C" void kernel_launch(void* const* d_in, const int* in_sizes, int n_in,
                              void* d_out, int out_size, void* d_ws, size_t ws_size,
                              hipStream_t stream) {
  const float* inputs = (const float*)d_in[0];
  const float* m1w0 = (const float*)d_in[1];  const float* m1b0 = (const float*)d_in[2];
  const float* m1w1 = (const float*)d_in[3];  const float* m1b1 = (const float*)d_in[4];
  const float* m2w0 = (const float*)d_in[5];  const float* m2b0 = (const float*)d_in[6];
  const float* m2w1 = (const float*)d_in[7];  const float* m2b1 = (const float*)d_in[8];
  const float* m3w0 = (const float*)d_in[9];  const float* m3b0 = (const float*)d_in[10];
  const float* m3w1 = (const float*)d_in[11]; const float* m3b1 = (const float*)d_in[12];

  // Workspace (~206 MB fp32). `big` time-shares: total_f [256,P] ->
  // per-batch attn [2048,2048] -> mlp3 hidden [512,P] (disjoint lifetimes).
  float* ws   = (float*)d_ws;
  float* x0   = ws; ws += 3   * PTS;
  float* lf   = ws; ws += 128 * PTS;   // local_f (xyz input to every SA layer)
  float* feat = ws; ws += 128 * PTS;
  float* xin  = ws; ws += 128 * PTS;   // also mlp hidden scratch
  float* q    = ws; ws += 32  * PTS;
  float* v    = ws; ws += 128 * PTS;
  float* xr   = ws; ws += 128 * PTS;
  float* lcat = ws; ws += 384 * PTS;   // l1|l2|l3 contiguous => concat is free
  float* big  = ws; ws += 512 * PTS;
  float* gmax = ws; ws += 2048;
  float* cs   = ws; ws += 2048;
  float* out  = (float*)d_out;

  // Stage 1: input transpose + mlp1 -> local_f
  transpose_in_k<<<PTS / 256, 256, 0, stream>>>(inputs, x0);
  gemm<1, false, false, true>(stream, m1w0, 3, x0, PTS, nullptr, xin, PTS,
                              128, 3, PTS, m1b0);
  gemm<0>(stream, m1w1, 128, xin, PTS, nullptr, lf, PTS, 128, 128, PTS, m1b1);

  // Stage 2: global max feature, total_f = [local_f ; broadcast(max)], mlp2 -> feat
  rowmax_k<<<2048 / 8, 256, 0, stream>>>(lf, gmax);
  build_totalf_k<<<dim3(PTS / 256, 128), 256, 0, stream>>>(lf, gmax, big);
  gemm<1>(stream, m2w0, 256, big, PTS, nullptr, xin,  PTS, 128, 256, PTS, m2b0);
  gemm<0>(stream, m2w1, 128, xin, PTS, nullptr, feat, PTS, 128, 128, PTS, m2b1);

  // Stage 3: three offset-attention layers
  const float* src = feat;
  for (int L = 0; L < 3; ++L) {
    const float* wq  = (const float*)d_in[13 + L * 7 + 0];
    const float* wv  = (const float*)d_in[13 + L * 7 + 1];
    const float* bvv = (const float*)d_in[13 + L * 7 + 2];
    const float* wt  = (const float*)d_in[13 + L * 7 + 3];
    const float* bt  = (const float*)d_in[13 + L * 7 + 4];
    const float* gg  = (const float*)d_in[13 + L * 7 + 5];
    const float* bbe = (const float*)d_in[13 + L * 7 + 6];
    float* lout = lcat + (size_t)L * 128 * PTS;

    add_k<<<dim3(PTS / 256, 128), 256, 0, stream>>>(src, lf, xin);  // x = x + xyz
    gemm<0>(stream, wq, 128, xin, PTS, nullptr, q, PTS, 32,  128, PTS, nullptr);
    gemm<0>(stream, wv, 128, xin, PTS, nullptr, v, PTS, 128, 128, PTS, bvv);

    for (int b = 0; b < NBAT; ++b) {
      const float* qb = q + b * NPTS;
      // energy[n][m] = sum_c q[c][n]*q[c][m]  (transA path; K=32 -> 1 WMMA/tile)
      gemm<0, true>(stream, qb, PTS, qb, PTS, nullptr, big, NPTS,
                    NPTS, 32, NPTS, nullptr);
      softmax_rows_k<<<NPTS, 256, 0, stream>>>(big);
      colsum_k<<<NPTS / 256, 256, 0, stream>>>(big, cs);
      // x_r[c][m] = (sum_n v[c][n]*attn[n][m]) / (1e-9 + colsum[m])
      gemm<3>(stream, v + b * NPTS, PTS, big, NPTS, nullptr, xr + b * NPTS, PTS,
              128, NPTS, NPTS, nullptr, nullptr, nullptr, nullptr, cs);
    }
    // l = xin + relu(BN(wt @ (xin - x_r) + bt))  (Bsub fuses the subtraction)
    gemm<2, false, true>(stream, wt, 128, xin, PTS, xr, lout, PTS,
                         128, 128, PTS, bt, gg, bbe, xin);
    src = lout;
  }

  // Stage 4: mlp3 + fused max-over-N (never materialize [B,1024,N])
  gemm<1>(stream, m3w0, 384, lcat, PTS, nullptr, big, PTS, 512, 384, PTS, m3b0);
  init_out_k<<<out_size / 256, 256, 0, stream>>>(out);
  gemm<4>(stream, m3w1, 512, big, PTS, nullptr, nullptr, PTS, 1024, 512, PTS, m3b1,
          nullptr, nullptr, nullptr, nullptr, out, NPTS);
}
